// diffusers_CrossAttention_51299089383685
// MI455X (gfx1250) — compile-verified, matrix operand order checked by reading, not run on hardware
//
#include <hip/hip_runtime.h>

// ---------------------------------------------------------------------------
// diffusers CrossAttention (self-attn), B=4 S=2048 D=1024 H=16 dh=64, fp32 io.
// bf16 WMMA everywhere; flash-attention streaming softmax; K/V tiles staged
// into LDS by the Tensor Data Mover (tensor_load_to_lds, double-buffered,
// s_wait_tensorcnt) with TDM padding for conflict-free LDS fragment reads.
//   k1: Q = X Wq   -> bf16 [B,H,S,d]
//   k1: K = X Wk   -> bf16 [B,H,S,d]
//   k1: V = X Wv   -> bf16 [B,H,d,S]      (transposed: PV B-operand contiguous)
//   k2: flash attn -> bf16 [B*S, D]
//   k3: out = O Wo + bo -> fp32 d_out
// ---------------------------------------------------------------------------

#define HQ      16
#define DHEAD   64
#define BATCH   4
#define SEQ     2048
#define DMODEL  1024
#define MTOT    (BATCH * SEQ)      // 8192
#define ATT_SCALE 0.125f           // 64^-0.5

typedef __attribute__((ext_vector_type(16))) __bf16 v16bf;
typedef __attribute__((ext_vector_type(8)))  __bf16 v8bf;
typedef __attribute__((ext_vector_type(4)))  __bf16 v4bf;
typedef __attribute__((ext_vector_type(8)))  float  v8f;
typedef __attribute__((ext_vector_type(4)))  unsigned int v4u;
typedef __attribute__((ext_vector_type(8)))  int    v8i;
typedef __attribute__((ext_vector_type(4)))  int    v4i;

// A/B fragment loader (16x32 bf16, K-striped per ISA 7.12.2):
//   lane<16 : row = lane,    elements 0..7 = K0..7,  8..15 = K16..23
//   lane>=16: row = lane-16, elements 0..7 = K8..15, 8..15 = K24..31
__device__ __forceinline__ v16bf frag_ld(const __bf16* base, int ld, int lane) {
  const int r  = lane & 15;
  const int hf = lane >> 4;
  const __bf16* p = base + (size_t)r * ld + hf * 8;
  v8bf lo = *(const v8bf*)(p);
  v8bf hi = *(const v8bf*)(p + 16);
  return __builtin_shufflevector(lo, hi, 0,1,2,3,4,5,6,7,8,9,10,11,12,13,14,15);
}

__device__ __forceinline__ v8f wmma_bf16(v16bf a, v16bf b, v8f c) {
  return __builtin_amdgcn_wmma_f32_16x16x32_bf16(false, a, false, b, (short)0, c,
                                                 false, false);
}

// ---------------------------------------------------------------------------
// TDM: 2D tile load (64 x 64 bf16) global -> LDS, with LDS padding of
// 4 DWORDs every 32 DWORDs (row stride 64 -> 72 elements).
// D# packing per CDNA5 ISA 8.3/8.4.
// ---------------------------------------------------------------------------
__device__ __forceinline__ unsigned lds_off32(const void* p) {
  // flat LDS aperture: addr[31:0] == LDS byte offset
  return (unsigned)(unsigned long long)p;
}

__device__ __forceinline__ void tdm_load_tile64(const __bf16* gsrc, unsigned ldsbyte,
                                                unsigned tensor_d0, unsigned tensor_d1,
                                                unsigned stride0) {
  const unsigned long long ga = (unsigned long long)gsrc;
  v4u g0;
  g0[0] = 1u;                                        // count=1, user-mode, no gather
  g0[1] = ldsbyte;                                   // lds_addr
  g0[2] = (unsigned)(ga & 0xffffffffu);              // global_addr[31:0]
  g0[3] = (unsigned)((ga >> 32) & 0x01ffffffu) | 0x80000000u; // addr[56:32] | type=2
  v8i g1;
  // data_size=2B (code 1), pad_enable, pad_interval=32DW (code 4),
  // pad_amount=4DW (code 3)
  g1[0] = (int)((1u << 16) | (1u << 20) | (4u << 22) | (3u << 25));
  g1[1] = (int)((tensor_d0 & 0xffffu) << 16);                       // dim0[15:0]
  g1[2] = (int)((tensor_d0 >> 16) | ((tensor_d1 & 0xffffu) << 16)); // dim0 hi, dim1 lo
  g1[3] = (int)((tensor_d1 >> 16) | (64u << 16));                   // dim1 hi, tile_dim0=64
  g1[4] = (int)64;                                                  // tile_dim1=64, tile_dim2=0
  g1[5] = (int)stride0;                                             // dim0_stride[31:0]
  g1[6] = 0;                                                        // stride hi / dim1_stride
  g1[7] = 0;
  const v4i z4 = {0, 0, 0, 0};
#if __clang_major__ >= 23
  const v8i z8 = {0, 0, 0, 0, 0, 0, 0, 0};
  __builtin_amdgcn_tensor_load_to_lds(g0, g1, z4, z4, z8, 0);
#else
  __builtin_amdgcn_tensor_load_to_lds(g0, g1, z4, z4, 0);
#endif
}

// ---------------------------------------------------------------------------
// GEMM: C[8192,1024] = X[8192,1024] * W[1024,1024]
// block = 256 thr (8 waves, 2x4), tile 128x128, k-step 32, 8 WMMA/wave/step.
// ---------------------------------------------------------------------------
template <int IN_BF16, int OUT_MODE>
__global__ __launch_bounds__(256) void gemm128(const void* __restrict__ Xv,
                                               const float* __restrict__ W,
                                               const float* __restrict__ bias,
                                               void* __restrict__ outv) {
  __shared__ __bf16 As[128 * 40];   // X tile, row-major, padded stride 40
  __shared__ __bf16 Bs[128 * 40];   // W tile transposed: Bs[n][k]
  const int tid  = threadIdx.x;
  const int lane = tid & 31;
  const int wave = tid >> 5;
  const int wm   = wave & 1;
  const int wn   = wave >> 1;
  const int n0   = blockIdx.x * 128;
  const int m0   = blockIdx.y * 128;

  const v8f vzero = {0.f,0.f,0.f,0.f,0.f,0.f,0.f,0.f};
  v8f acc[4][2];
#pragma unroll
  for (int i = 0; i < 4; ++i)
#pragma unroll
    for (int j = 0; j < 2; ++j) acc[i][j] = vzero;

  for (int k0 = 0; k0 < DMODEL; k0 += 32) {
    if (k0 + 32 < DMODEL) {
      const int esz = IN_BF16 ? 2 : 4;
      __builtin_prefetch((const char*)Xv +
          ((size_t)(m0 + (tid & 127)) * DMODEL + k0 + 32 + (tid >> 7) * 16) * esz, 0, 0);
      __builtin_prefetch((const char*)W +
          ((size_t)(k0 + 32 + (tid >> 3)) * DMODEL + n0 + (tid & 7) * 16) * 4, 0, 0);
    }
#pragma unroll
    for (int i = tid; i < 128 * 8; i += 256) {
      const int r = i >> 3, c4 = (i & 7) << 2;
      __bf16* d = &As[r * 40 + c4];
      if (IN_BF16) {
        v4bf x = *(const v4bf*)((const __bf16*)Xv + (size_t)(m0 + r) * DMODEL + k0 + c4);
        d[0] = x[0]; d[1] = x[1]; d[2] = x[2]; d[3] = x[3];
      } else {
        const float4 x = *(const float4*)((const float*)Xv + (size_t)(m0 + r) * DMODEL + k0 + c4);
        d[0] = (__bf16)x.x; d[1] = (__bf16)x.y; d[2] = (__bf16)x.z; d[3] = (__bf16)x.w;
      }
    }
#pragma unroll
    for (int i = tid; i < 32 * 32; i += 256) {
      const int kr = i >> 5, c4 = (i & 31) << 2;
      const float4 w = *(const float4*)(W + (size_t)(k0 + kr) * DMODEL + n0 + c4);
      Bs[(c4 + 0) * 40 + kr] = (__bf16)w.x;
      Bs[(c4 + 1) * 40 + kr] = (__bf16)w.y;
      Bs[(c4 + 2) * 40 + kr] = (__bf16)w.z;
      Bs[(c4 + 3) * 40 + kr] = (__bf16)w.w;
    }
    __syncthreads();

    v16bf af[4], bf[2];
#pragma unroll
    for (int i = 0; i < 4; ++i) af[i] = frag_ld(&As[(wm * 64 + i * 16) * 40], 40, lane);
#pragma unroll
    for (int j = 0; j < 2; ++j) bf[j] = frag_ld(&Bs[(wn * 32 + j * 16) * 40], 40, lane);
#pragma unroll
    for (int i = 0; i < 4; ++i)
#pragma unroll
      for (int j = 0; j < 2; ++j) acc[i][j] = wmma_bf16(af[i], bf[j], acc[i][j]);
    __syncthreads();
  }

  const int r15 = lane & 15, hf = lane >> 4;
#pragma unroll
  for (int i = 0; i < 4; ++i)
#pragma unroll
    for (int j = 0; j < 2; ++j) {
      const v8f c = acc[i][j];
#pragma unroll
      for (int r = 0; r < 8; ++r) {
        const int m = m0 + wm * 64 + i * 16 + hf * 8 + r;
        const int n = n0 + wn * 32 + j * 16 + r15;
        const float v = c[r];
        if (OUT_MODE == 2) {
          ((float*)outv)[(size_t)m * DMODEL + n] = v + bias[n];
        } else {
          const int b = m >> 11, s = m & (SEQ - 1);
          const int h = n >> 6, dh = n & (DHEAD - 1);
          size_t idx;
          if (OUT_MODE == 0)
            idx = ((size_t)(b * HQ + h) * SEQ + s) * DHEAD + dh;   // [B,H,S,d]
          else
            idx = ((size_t)(b * HQ + h) * DHEAD + dh) * SEQ + s;   // [B,H,d,S]
          ((__bf16*)outv)[idx] = (__bf16)v;
        }
      }
    }
}

// ---------------------------------------------------------------------------
// Flash attention. grid = (S/64, B*H); block = 128 thr (4 waves).
// Each wave: 16 queries. Keys streamed in blocks of 64 via TDM double
// buffering: wave0 issues tensor_load_to_lds for K (64keys x 64d) and V^T
// (64dv x 64keys), waits on TENSORcnt, barrier, then all waves run
// 8 WMMA (scores) + 8 WMMA (PV) per block out of padded LDS (stride 72).
// ---------------------------------------------------------------------------
__global__ __launch_bounds__(128) void attn_flash(const __bf16* __restrict__ Qh,
                                                  const __bf16* __restrict__ Kh,
                                                  const __bf16* __restrict__ Vt,
                                                  __bf16* __restrict__ Ob) {
  __shared__ __bf16 Kt[2][64 * 72];     // keys x d, padded
  __shared__ __bf16 Vs[2][64 * 72];     // dv x keys, padded
  __shared__ __bf16 Plds[4][16 * 72];   // per-wave P staging
  const int tid  = threadIdx.x;
  const int lane = tid & 31;
  const int wave = tid >> 5;
  const int r15  = lane & 15, hf = lane >> 4;
  const int bh   = blockIdx.y;                 // b*H + h
  const int q0   = blockIdx.x * 64 + wave * 16;

  const __bf16* Qb = Qh + ((size_t)bh * SEQ + q0) * DHEAD;
  const __bf16* Kb = Kh + (size_t)bh * SEQ * DHEAD;
  const __bf16* Vb = Vt + (size_t)bh * DHEAD * SEQ;

  v16bf qa[2];
#pragma unroll
  for (int f = 0; f < 2; ++f) qa[f] = frag_ld(Qb + f * 32, DHEAD, lane);

  const v8f vzero = {0.f,0.f,0.f,0.f,0.f,0.f,0.f,0.f};
  v8f o[4];
#pragma unroll
  for (int t = 0; t < 4; ++t) o[t] = vzero;
  float mrow[8], lrow[8];
#pragma unroll
  for (int r = 0; r < 8; ++r) { mrow[r] = -3.0e38f; lrow[r] = 0.f; }

  const int NKB = SEQ / 64;             // 32 key blocks

  if (wave == 0) {                      // prologue: block 0 -> buffer 0
    tdm_load_tile64(Kb, lds_off32(&Kt[0][0]), DHEAD, SEQ, DHEAD);
    tdm_load_tile64(Vb, lds_off32(&Vs[0][0]), SEQ, DHEAD, SEQ);
  }

  for (int n = 0; n < NKB; ++n) {
    const int p = n & 1;
    __syncthreads();                    // buffer p^1 readers from iter n-1 done
    if (wave == 0) {
      if (n + 1 < NKB) {
        const int kb1 = (n + 1) * 64;
        tdm_load_tile64(Kb + (size_t)kb1 * DHEAD, lds_off32(&Kt[p ^ 1][0]),
                        DHEAD, SEQ, DHEAD);
        tdm_load_tile64(Vb + kb1, lds_off32(&Vs[p ^ 1][0]),
                        SEQ, DHEAD, SEQ);
        __builtin_amdgcn_s_wait_tensorcnt(2);   // block n's 2 loads complete
      } else {
        __builtin_amdgcn_s_wait_tensorcnt(0);
      }
    }
    __syncthreads();                    // buffer p visible to all waves

    // ---- scores: 16q x 64k = 4 subtiles x 2 k-steps = 8 WMMA
    v8f s[4];
#pragma unroll
    for (int t = 0; t < 4; ++t) {
      const __bf16* kbase = &Kt[p][(t * 16) * 72];
      v8f acc = vzero;
      acc = wmma_bf16(qa[0], frag_ld(kbase + 0,  72, lane), acc);
      acc = wmma_bf16(qa[1], frag_ld(kbase + 32, 72, lane), acc);
      s[t] = acc;
    }

    // ---- online softmax over 64 keys
    float pr[4][8];
#pragma unroll
    for (int r = 0; r < 8; ++r) {
      float x0 = s[0][r] * ATT_SCALE, x1 = s[1][r] * ATT_SCALE;
      float x2 = s[2][r] * ATT_SCALE, x3 = s[3][r] * ATT_SCALE;
      float tm = fmaxf(fmaxf(x0, x1), fmaxf(x2, x3));
      tm = fmaxf(tm, __shfl_xor(tm, 1, 32));
      tm = fmaxf(tm, __shfl_xor(tm, 2, 32));
      tm = fmaxf(tm, __shfl_xor(tm, 4, 32));
      tm = fmaxf(tm, __shfl_xor(tm, 8, 32));
      const float mnew  = fmaxf(mrow[r], tm);
      const float alpha = __expf(mrow[r] - mnew);
      mrow[r] = mnew;
      const float e0 = __expf(x0 - mnew), e1 = __expf(x1 - mnew);
      const float e2 = __expf(x2 - mnew), e3 = __expf(x3 - mnew);
      pr[0][r] = e0; pr[1][r] = e1; pr[2][r] = e2; pr[3][r] = e3;
      float ts = (e0 + e1) + (e2 + e3);
      ts += __shfl_xor(ts, 1, 32);
      ts += __shfl_xor(ts, 2, 32);
      ts += __shfl_xor(ts, 4, 32);
      ts += __shfl_xor(ts, 8, 32);
      lrow[r] = lrow[r] * alpha + ts;
      o[0][r] *= alpha; o[1][r] *= alpha; o[2][r] *= alpha; o[3][r] *= alpha;
    }

    // ---- P: C-layout -> wave-private LDS -> two A-layout fragments
    __bf16* pw = &Plds[wave][0];
#pragma unroll
    for (int r = 0; r < 8; ++r) {
      const int row = hf * 8 + r;
#pragma unroll
      for (int t = 0; t < 4; ++t)
        pw[row * 72 + t * 16 + r15] = (__bf16)pr[t][r];
    }
    const v16bf pa0 = frag_ld(pw + 0,  72, lane);   // keys 0..31
    const v16bf pa1 = frag_ld(pw + 32, 72, lane);   // keys 32..63

    // ---- O += P V : 4 dv subtiles x 2 k-steps = 8 WMMA
#pragma unroll
    for (int t = 0; t < 4; ++t) {
      const __bf16* vbase = &Vs[p][(t * 16) * 72];
      o[t] = wmma_bf16(pa0, frag_ld(vbase + 0,  72, lane), o[t]);
      o[t] = wmma_bf16(pa1, frag_ld(vbase + 32, 72, lane), o[t]);
    }
  }

  // normalize, store bf16 [B*S, D]
  const int b = bh >> 4, h = bh & 15;
#pragma unroll
  for (int r = 0; r < 8; ++r) {
    const float inv = 1.f / lrow[r];
    const size_t m = (size_t)b * SEQ + q0 + hf * 8 + r;
#pragma unroll
    for (int t = 0; t < 4; ++t) {
      const int n = h * 64 + t * 16 + r15;
      Ob[m * DMODEL + n] = (__bf16)(o[t][r] * inv);
    }
  }
}

// ---------------------------------------------------------------------------
extern "C" void kernel_launch(void* const* d_in, const int* in_sizes, int n_in,
                              void* d_out, int out_size, void* d_ws, size_t ws_size,
                              hipStream_t stream) {
  const float* X  = (const float*)d_in[0];
  const float* Wq = (const float*)d_in[1];
  const float* Wk = (const float*)d_in[2];
  const float* Wv = (const float*)d_in[3];
  const float* Wo = (const float*)d_in[4];
  const float* bo = (const float*)d_in[5];

  const size_t NELT = (size_t)MTOT * DMODEL;   // 8M elements, 16MB bf16 each
  __bf16* Qh = (__bf16*)d_ws;
  __bf16* Kh = Qh + NELT;
  __bf16* Vt = Kh + NELT;
  __bf16* Ob = Vt + NELT;

  const dim3 gg(DMODEL / 128, MTOT / 128);     // (8, 64)
  gemm128<0, 0><<<gg, 256, 0, stream>>>(X, Wq, nullptr, Qh);
  gemm128<0, 0><<<gg, 256, 0, stream>>>(X, Wk, nullptr, Kh);
  gemm128<0, 1><<<gg, 256, 0, stream>>>(X, Wv, nullptr, Vt);

  attn_flash<<<dim3(SEQ / 64, BATCH * HQ), 128, 0, stream>>>(Qh, Kh, Vt, Ob);

  gemm128<1, 2><<<gg, 256, 0, stream>>>(Ob, Wo, bo, d_out);
}
